// GAT_44461501448667
// MI455X (gfx1250) — compile-verified, matrix operand order checked by reading, not run on hardware
//
#include <hip/hip_runtime.h>
#include <math.h>

typedef __attribute__((ext_vector_type(16))) _Float16 v16h;
typedef __attribute__((ext_vector_type(8)))  _Float16 v8h;
typedef __attribute__((ext_vector_type(8)))  float    v8f;

#define TB 256
#define BM 128
#define BN 64
#define BK 32
#define AP 40   // LDS pitch in halves (pad vs 32 to spread banks)

// ---- CDNA5 async global->LDS path (guarded; falls back to sync copies) ----
#if defined(__HIP_DEVICE_COMPILE__) && \
    __has_builtin(__builtin_amdgcn_global_load_async_to_lds_b128) && \
    __has_builtin(__builtin_amdgcn_s_wait_asynccnt)
#define USE_ASYNC 1
#else
#define USE_ASYNC 0
#endif

#if USE_ASYNC
typedef int v4i_vs __attribute__((vector_size(16)));
typedef __attribute__((address_space(1))) v4i_vs* gl_v4i_ptr;   // global int4*
typedef __attribute__((address_space(3))) v4i_vs* ld_v4i_ptr;   // LDS int4*
__device__ __forceinline__ void cp16(const _Float16* g, _Float16* l){
  // AS1 addr == flat addr for global; AS3 32-bit offset == flat addr[31:0] (ISA aperture rule)
  __builtin_amdgcn_global_load_async_to_lds_b128(
      (gl_v4i_ptr)(uintptr_t)g, (ld_v4i_ptr)(uintptr_t)l, 0, 0);
}
#else
__device__ __forceinline__ void cp16(const _Float16* g, _Float16* l){
  *(v8h*)l = *(const v8h*)g;
}
#endif

static inline size_t cdiv_sz(size_t a, size_t b){ return (a + b - 1) / b; }

// ------------------------------------------------------------------
// f32 -> f16 activation convert (zero-pads rows >= N)
// ------------------------------------------------------------------
__global__ void cvt_x_kernel(const float* __restrict__ x, _Float16* __restrict__ out,
                             int N, int K, size_t total){
  size_t i = (size_t)blockIdx.x * TB + threadIdx.x;
  if (i >= total) return;
  int row = (int)(i / (size_t)K);
  out[i] = (row < N) ? (_Float16)x[i] : (_Float16)0.f;
}

// concat [W | lw] -> f16 weight buffer, TRANSPOSED to (Mg+Ml) x K row-major
// so the WMMA B operand needs per-lane K-contiguous chunks only (no LDS transpose)
__global__ void cvt_w_kernel(const float* __restrict__ W, const float* __restrict__ lw,
                             _Float16* __restrict__ out, int K, int Mg, int Ml){
  size_t total = (size_t)K * (Mg + Ml);
  size_t i = (size_t)blockIdx.x * TB + threadIdx.x;   // i = c*K + k
  if (i >= total) return;
  int c = (int)(i / (size_t)K);
  int k = (int)(i - (size_t)c * K);
  float v = (c < Mg) ? W[(size_t)k * Mg + c] : lw[(size_t)k * Ml + (c - Mg)];
  out[i] = (_Float16)v;
}

__global__ void fill_f32_kernel(float* __restrict__ p, float v, size_t n){
  size_t i = (size_t)blockIdx.x * TB + threadIdx.x;
  if (i < n) p[i] = v;
}

// ------------------------------------------------------------------
// WMMA GEMM: C[Npad x M] = A[Npad x K] (f16 row-major) * Bt[M x K] (f16, pre-transposed)
// block tile 128x64, 8 waves, wave tile 16x64, K-step 32,
// double-buffered LDS with async global->LDS prefetch of step s+1 under step s WMMAs
// ------------------------------------------------------------------
__global__ __launch_bounds__(TB) void gemm_wmma_kernel(const _Float16* __restrict__ A,
                                                       const _Float16* __restrict__ Bt,
                                                       float* __restrict__ C,
                                                       int K, int M){
  __shared__ _Float16 As[2][BM * AP];
  __shared__ _Float16 Bs[2][BN * AP];

  const int tid  = threadIdx.x;
  const int wave = tid >> 5;
  const int lane = tid & 31;
  const int row0 = blockIdx.y * BM;
  const int col0 = blockIdx.x * BN;

  // staging coordinates (fixed per thread; k0 added as we go)
  const int ar   = tid >> 1;            // A row 0..127, 16 halves per thread
  const int aseg = (tid & 1) * 16;
  const _Float16* agp = A + (size_t)(row0 + ar) * K + aseg;
  const int br   = tid >> 2;            // B row (output col) 0..63, 8 halves per thread
  const int bseg = (tid & 3) * 8;
  const _Float16* bgp = Bt + (size_t)(col0 + br) * K + bseg;

  v8f acc[4] = {};

  // WMMA 16-bit operand lane mapping (ISA 7.12.2):
  // lanes 0-15: K chunks {0..7,16..23}; lanes 16-31: K chunks {8..15,24..31}
  const int m_l = (lane & 15) + wave * 16;
  const int kb  = (lane < 16) ? 0 : 8;
  const int nsteps = K / BK;

  // prologue: stage step 0 into buffer 0
  cp16(agp,     &As[0][ar * AP + aseg]);
  cp16(agp + 8, &As[0][ar * AP + aseg + 8]);
  cp16(bgp,     &Bs[0][br * AP + bseg]);

  int buf = 0;
  for (int s = 0; s < nsteps; ++s) {
    if (s + 1 < nsteps) {
      const int k0 = (s + 1) * BK;
      cp16(agp + k0,     &As[buf ^ 1][ar * AP + aseg]);
      cp16(agp + k0 + 8, &As[buf ^ 1][ar * AP + aseg + 8]);
      cp16(bgp + k0,     &Bs[buf ^ 1][br * AP + bseg]);
#if USE_ASYNC
      __builtin_amdgcn_s_wait_asynccnt(3);   // 3 newest in flight; current buf landed
#endif
    } else {
#if USE_ASYNC
      __builtin_amdgcn_s_wait_asynccnt(0);
#endif
    }
    __syncthreads();   // all waves' writes to As/Bs[buf] complete

    v16h a;
    {
      v8h lo = *(const v8h*)&As[buf][m_l * AP + kb];
      v8h hi = *(const v8h*)&As[buf][m_l * AP + kb + 16];
      a = __builtin_shufflevector(lo, hi, 0,1,2,3,4,5,6,7,8,9,10,11,12,13,14,15);
    }
#pragma unroll
    for (int j = 0; j < 4; ++j) {
      int n_l = (lane & 15) + j * 16;
      v8h lo = *(const v8h*)&Bs[buf][n_l * AP + kb];
      v8h hi = *(const v8h*)&Bs[buf][n_l * AP + kb + 16];
      v16h b = __builtin_shufflevector(lo, hi, 0,1,2,3,4,5,6,7,8,9,10,11,12,13,14,15);
      acc[j] = __builtin_amdgcn_wmma_f32_16x16x32_f16(false, a, false, b,
                                                      (short)0, acc[j], false, false);
    }
    __syncthreads();   // frees As/Bs[buf] for the prefetch issued next iteration
    buf ^= 1;
  }

  // store C (f32 16x16 layout: VGPR r -> M=r (lanes 0-15) / r+8 (lanes 16-31))
  const int rbase = row0 + wave * 16 + ((lane >> 4) << 3);
  const int cbase = col0 + (lane & 15);
#pragma unroll
  for (int j = 0; j < 4; ++j)
#pragma unroll
    for (int r = 0; r < 8; ++r)
      C[(size_t)(rbase + r) * M + cbase + j * 16] = acc[j][r];
}

// ------------------------------------------------------------------
// per-node attention logits: alpha_s/d[n,h] = sum_c hW[n, h*256+c] * a[h,c]
// ------------------------------------------------------------------
__global__ void alpha_kernel(const float* __restrict__ hW, int pitch,
                             const float* __restrict__ asrc, const float* __restrict__ adst,
                             float* __restrict__ out_s, float* __restrict__ out_d,
                             int N, int H){
  int gw   = (blockIdx.x * TB + threadIdx.x) >> 5;
  int lane = threadIdx.x & 31;
  if (gw >= N * H) return;
  int n = gw / H, h = gw - n * H;
  const float* hp = hW   + (size_t)n * pitch + h * 256;
  const float* sp = asrc + h * 256;
  const float* dp = adst + h * 256;
  float s = 0.f, d = 0.f;
  for (int c = lane; c < 256; c += 32) { float v = hp[c]; s += v * sp[c]; d += v * dp[c]; }
  for (int o = 16; o > 0; o >>= 1) { s += __shfl_down(s, o, 32); d += __shfl_down(d, o, 32); }
  if (lane == 0) { out_s[gw] = s; out_d[gw] = d; }
}

__device__ inline void edge_sd(const int* __restrict__ ei, int i, int E, int& s, int& d){
  if (i < E) { s = ei[i]; d = ei[E + i]; } else { s = i - E; d = s; }  // self loops
}

// monotone float atomic-max via int/uint ordering trick
__device__ inline void atomicMaxF(float* a, float v){
  if (v >= 0.f) atomicMax((int*)a, __float_as_int(v));
  else          atomicMin((unsigned int*)a, __float_as_uint(v));
}

__global__ void edge_max_kernel(const int* __restrict__ ei,
                                const float* __restrict__ as, const float* __restrict__ ad,
                                float* __restrict__ ebuf, float* __restrict__ mbuf,
                                int E, int Etot, int H){
  int idx = blockIdx.x * TB + threadIdx.x;
  if (idx >= Etot * H) return;
  int i = idx / H, h = idx - i * H;
  int s, d; edge_sd(ei, i, E, s, d);
  float v = as[s * H + h] + ad[d * H + h];
  v = (v > 0.f) ? v : 0.2f * v;                 // leaky_relu(0.2)
  ebuf[idx] = v;
  atomicMaxF(&mbuf[d * H + h], v);
}

__global__ void edge_exp_kernel(const int* __restrict__ ei,
                                float* __restrict__ ebuf, const float* __restrict__ mbuf,
                                float* __restrict__ denom, int E, int Etot, int H){
  int idx = blockIdx.x * TB + threadIdx.x;
  if (idx >= Etot * H) return;
  int i = idx / H, h = idx - i * H;
  int s, d; edge_sd(ei, i, E, s, d);
  float w = __expf(ebuf[idx] - mbuf[d * H + h]);
  ebuf[idx] = w;
  atomicAdd(&denom[d * H + h], w);
}

// msg scatter, 4 channels per thread: agg[dst, j..j+3] += hW[src, j..j+3] * alpha
__global__ void scatter_kernel(const int* __restrict__ ei, const float* __restrict__ hW, int pitch,
                               const float* __restrict__ ebuf, const float* __restrict__ denom,
                               float* __restrict__ agg, int E, int H, int HC, size_t total4){
  size_t t = (size_t)blockIdx.x * TB + threadIdx.x;
  if (t >= total4) return;
  const int HC4 = HC >> 2;
  int i = (int)(t / (size_t)HC4);
  int q = (int)(t - (size_t)i * HC4);
  int j = q << 2;
  int h = j >> 8;                                // C = 256
  int s, d; edge_sd(ei, i, E, s, d);
  float a = ebuf[(size_t)i * H + h] / (denom[d * H + h] + 1e-16f);
  float4 v = *(const float4*)(hW + (size_t)s * pitch + j);
  float* ap = agg + (size_t)d * HC + j;
  atomicAdd(ap + 0, v.x * a);
  atomicAdd(ap + 1, v.y * a);
  atomicAdd(ap + 2, v.z * a);
  atomicAdd(ap + 3, v.w * a);
}

// layers 1-2: elu(agg + b + lin + lb) -> f16 (next GEMM A operand)
__global__ void finalize_kernel(const float* __restrict__ agg, const float* __restrict__ Cg,
                                const float* __restrict__ b, const float* __restrict__ lb,
                                _Float16* __restrict__ out16, int HC, int Mtot, int Mgat,
                                size_t total){
  size_t i = (size_t)blockIdx.x * TB + threadIdx.x;
  if (i >= total) return;
  int n = (int)(i / (size_t)HC);
  int j = (int)(i - (size_t)n * HC);
  float v = agg[i] + b[j] + Cg[(size_t)n * Mtot + Mgat + j] + lb[j];
  v = (v > 0.f) ? v : (__expf(v) - 1.f);        // elu
  out16[i] = (_Float16)v;
}

// layer 3: mean over 6 heads + b3 + lin + lb3 -> fp32 output
__global__ void output_kernel(const float* __restrict__ agg, const float* __restrict__ Cg,
                              const float* __restrict__ b, const float* __restrict__ lb,
                              float* __restrict__ out, size_t total){
  size_t i = (size_t)blockIdx.x * TB + threadIdx.x;
  if (i >= total) return;
  int n = (int)(i / 256);
  int c = (int)(i - (size_t)n * 256);
  float s = 0.f;
#pragma unroll
  for (int h = 0; h < 6; ++h) s += agg[(size_t)n * 1536 + h * 256 + c];
  out[i] = s * (1.f / 6.f) + b[c] + Cg[(size_t)n * 1792 + 1536 + c] + lb[c];
}

// ------------------------------------------------------------------
// host-side per-layer GAT pipeline
// ------------------------------------------------------------------
static void run_gat(hipStream_t stream, const _Float16* A16, int K,
                    const float* W, const float* lw, int Mg, int Ml,
                    const float* a_s, const float* a_d, int H,
                    const int* ei, int E, int Etot, int N, int Npad,
                    _Float16* W16, float* Cg, float* agg,
                    float* ebuf, float* alphs, float* alphd, float* mbuf, float* denom){
  const int M  = Mg + Ml;
  const int HC = Mg;                 // H * 256
  cvt_w_kernel<<<cdiv_sz((size_t)K * M, TB), TB, 0, stream>>>(W, lw, W16, K, Mg, Ml);
  dim3 g(M / BN, Npad / BM);
  gemm_wmma_kernel<<<g, TB, 0, stream>>>(A16, W16, Cg, K, M);
  alpha_kernel<<<cdiv_sz((size_t)N * H * 32, TB), TB, 0, stream>>>(Cg, M, a_s, a_d, alphs, alphd, N, H);
  fill_f32_kernel<<<cdiv_sz((size_t)N * H, TB), TB, 0, stream>>>(mbuf, -INFINITY, (size_t)N * H);
  fill_f32_kernel<<<cdiv_sz((size_t)N * H, TB), TB, 0, stream>>>(denom, 0.f, (size_t)N * H);
  fill_f32_kernel<<<cdiv_sz((size_t)N * HC, TB), TB, 0, stream>>>(agg, 0.f, (size_t)N * HC);
  edge_max_kernel<<<cdiv_sz((size_t)Etot * H, TB), TB, 0, stream>>>(ei, alphs, alphd, ebuf, mbuf, E, Etot, H);
  edge_exp_kernel<<<cdiv_sz((size_t)Etot * H, TB), TB, 0, stream>>>(ei, ebuf, mbuf, denom, E, Etot, H);
  size_t stot4 = (size_t)Etot * (HC >> 2);
  scatter_kernel<<<cdiv_sz(stot4, TB), TB, 0, stream>>>(ei, Cg, M, ebuf, denom, agg, E, H, HC, stot4);
}

extern "C" void kernel_launch(void* const* d_in, const int* in_sizes, int n_in,
                              void* d_out, int out_size, void* d_ws, size_t ws_size,
                              hipStream_t stream){
  const float* x   = (const float*)d_in[0];
  const int*   ei  = (const int*)  d_in[1];
  const float* W1  = (const float*)d_in[3];
  const float* as1 = (const float*)d_in[4];
  const float* ad1 = (const float*)d_in[5];
  const float* b1  = (const float*)d_in[6];
  const float* lw1 = (const float*)d_in[7];
  const float* lb1 = (const float*)d_in[8];
  const float* W2  = (const float*)d_in[9];
  const float* as2 = (const float*)d_in[10];
  const float* ad2 = (const float*)d_in[11];
  const float* b2  = (const float*)d_in[12];
  const float* lw2 = (const float*)d_in[13];
  const float* lb2 = (const float*)d_in[14];
  const float* W3  = (const float*)d_in[15];
  const float* as3 = (const float*)d_in[16];
  const float* ad3 = (const float*)d_in[17];
  const float* b3  = (const float*)d_in[18];
  const float* lw3 = (const float*)d_in[19];
  const float* lb3 = (const float*)d_in[20];
  float* out = (float*)d_out;

  const int N    = in_sizes[0] / 512;
  const int E    = in_sizes[1] / 2;
  const int Etot = E + N;
  const int Npad = ((N + BM - 1) / BM) * BM;

  char* ws = (char*)d_ws;
  auto carve = [&](size_t bytes) -> void* {
    void* p = (void*)ws; ws += (bytes + 255) & ~(size_t)255; return p;
  };
  _Float16* A16x = (_Float16*)carve((size_t)Npad * 512  * 2);
  _Float16* A16h = (_Float16*)carve((size_t)Npad * 1024 * 2);
  _Float16* W16  = (_Float16*)carve((size_t)1024 * 2048 * 2);
  float* Cg    = (float*)carve((size_t)Npad * 2048 * 4);
  float* agg   = (float*)carve((size_t)Npad * 1536 * 4);
  float* ebuf  = (float*)carve((size_t)Etot * 6 * 4);
  float* alphs = (float*)carve((size_t)N * 6 * 4);
  float* alphd = (float*)carve((size_t)N * 6 * 4);
  float* mbuf  = (float*)carve((size_t)N * 6 * 4);
  float* denom = (float*)carve((size_t)N * 6 * 4);

  // ---- layer 1 ----
  {
    size_t tot = (size_t)Npad * 512;
    cvt_x_kernel<<<cdiv_sz(tot, TB), TB, 0, stream>>>(x, A16x, N, 512, tot);
  }
  run_gat(stream, A16x, 512, W1, lw1, 1024, 1024, as1, ad1, 4,
          ei, E, Etot, N, Npad, W16, Cg, agg, ebuf, alphs, alphd, mbuf, denom);
  finalize_kernel<<<cdiv_sz((size_t)N * 1024, TB), TB, 0, stream>>>(
      agg, Cg, b1, lb1, A16h, 1024, 2048, 1024, (size_t)N * 1024);

  // ---- layer 2 ----
  run_gat(stream, A16h, 1024, W2, lw2, 1024, 1024, as2, ad2, 4,
          ei, E, Etot, N, Npad, W16, Cg, agg, ebuf, alphs, alphd, mbuf, denom);
  finalize_kernel<<<cdiv_sz((size_t)N * 1024, TB), TB, 0, stream>>>(
      agg, Cg, b2, lb2, A16h, 1024, 2048, 1024, (size_t)N * 1024);

  // ---- layer 3 ----
  run_gat(stream, A16h, 1024, W3, lw3, 1536, 256, as3, ad3, 6,
          ei, E, Etot, N, Npad, W16, Cg, agg, ebuf, alphs, alphd, mbuf, denom);
  output_kernel<<<cdiv_sz((size_t)out_size, TB), TB, 0, stream>>>(
      agg, Cg, b3, lb3, out, (size_t)out_size);
}
